// ScaledAttention_51015621542204
// MI455X (gfx1250) — matrix operand order repaired and checked by reading, not compiled, not run on hardware
//
#include <hip/hip_runtime.h>

typedef _Float16 half_t;
typedef __attribute__((ext_vector_type(16))) _Float16 v16h;
typedef __attribute__((ext_vector_type(8)))  _Float16 v8h;
typedef __attribute__((ext_vector_type(8)))  float    v8f;

union AF { v16h v; v8h h[2]; };

static __device__ __forceinline__ v16h load_frag16(const half_t* p0, const half_t* p1) {
  AF f;
  f.h[0] = *(const v8h*)p0;
  f.h[1] = *(const v8h*)p1;
  return f.v;
}

// -----------------------------------------------------------------------------
// CDNA5 async global -> LDS copy: 16 bytes per lane, tracked by ASYNCcnt.
// LDS destination is the low 32 bits of the flat shared-memory address
// (gfx1250 LDS aperture keeps the byte offset in addr[31:0]).
// -----------------------------------------------------------------------------
static __device__ __forceinline__ void g2l_async_b128(half_t* lds, const half_t* g) {
  unsigned loff = (unsigned)(size_t)lds;
  asm volatile("global_load_async_to_lds_b128 %0, %1, off"
               :: "v"(loff), "v"(g)
               : "memory");
}
#define WAIT_ASYNC(n) asm volatile("s_wait_asynccnt %0" :: "i"(n) : "memory")
#define WAIT_DS0()    asm volatile("s_wait_dscnt 0" ::: "memory")

// -----------------------------------------------------------------------------
// fp32 -> fp16 conversion, 4 elements / thread
// -----------------------------------------------------------------------------
__global__ void f32_to_f16_4(const float* __restrict__ in, half_t* __restrict__ out, int n4) {
  int i = blockIdx.x * blockDim.x + threadIdx.x;
  if (i < n4) {
    const float4 v = ((const float4*)in)[i];
    union { half_t h[4]; unsigned long long u; } o;
    o.h[0] = (half_t)v.x; o.h[1] = (half_t)v.y;
    o.h[2] = (half_t)v.z; o.h[3] = (half_t)v.w;
    ((unsigned long long*)out)[i] = o.u;
  }
}

// -----------------------------------------------------------------------------
// C[M,N] = A[M,K] * W[N,K]^T + bias  (f16 in, f32 accumulate, f16 out)
// Block 128 threads = 4 waves -> 128x64 C tile (wave = 32x64).
// A/W tiles double-buffered in LDS via async global->LDS copies.
// -----------------------------------------------------------------------------
__global__ __launch_bounds__(128) void qkv_gemm_wmma(
    const half_t* __restrict__ A,     // [M,K]
    const half_t* __restrict__ W,     // [N,K]
    const float*  __restrict__ bias,  // [N]
    half_t* __restrict__ Cmat,        // [M,N] or [N,M] if transposeOut
    int M, int N, int K, int transposeOut)
{
  __shared__ __align__(16) half_t Atile[2][128][64];   // 32 KB
  __shared__ __align__(16) half_t Wtile[2][64][64];    // 16 KB

  const int t    = threadIdx.x;
  const int lane = t & 31;
  const int wave = t >> 5;
  const int r    = lane & 15;
  const int sub  = lane >> 4;
  const int mBlk = blockIdx.x * 128;
  const int nBase = blockIdx.y * 64;

  v8f acc[2][4] = {};

  // cooperative async tile loader: A 128x64 (8 chunks/thread), W 64x64 (4 chunks/thread)
  auto load_tiles = [&](int k0, int buf) {
    #pragma unroll
    for (int i = 0; i < 8; ++i) {
      const int c = t + 128 * i;
      const int row = c >> 3, col = (c & 7) * 8;
      g2l_async_b128(&Atile[buf][row][col],
                     A + (size_t)(mBlk + row) * K + k0 + col);
    }
    #pragma unroll
    for (int i = 0; i < 4; ++i) {
      const int c = t + 128 * i;
      const int row = c >> 3, col = (c & 7) * 8;
      g2l_async_b128(&Wtile[buf][row][col],
                     W + (size_t)(nBase + row) * K + k0 + col);
    }
  };

  load_tiles(0, 0);

  for (int k0 = 0; k0 < K; k0 += 64) {
    const int buf = (k0 >> 6) & 1;
    const bool hasNext = (k0 + 64 < K);
    if (hasNext) { load_tiles(k0 + 64, buf ^ 1); WAIT_ASYNC(12); }
    else         { WAIT_ASYNC(0); }
    __syncthreads();

    #pragma unroll
    for (int ks = 0; ks < 64; ks += 32) {
      // hoist ALL fragment loads into distinct registers, then 8 back-to-back wmma
      const half_t* ap0 = &Atile[buf][wave * 32 + r][ks + sub * 8];
      const half_t* ap1 = &Atile[buf][wave * 32 + 16 + r][ks + sub * 8];
      const v16h a0 = load_frag16(ap0, ap0 + 16);
      const v16h a1 = load_frag16(ap1, ap1 + 16);
      v16h bf[4];
      #pragma unroll
      for (int nf = 0; nf < 4; ++nf) {
        const half_t* wp = &Wtile[buf][nf * 16 + r][ks + sub * 8];
        bf[nf] = load_frag16(wp, wp + 16);
      }
      #pragma unroll
      for (int nf = 0; nf < 4; ++nf) {
        acc[0][nf] = __builtin_amdgcn_wmma_f32_16x16x32_f16(
            false, a0, false, bf[nf], (short)0, acc[0][nf], false, false);
        acc[1][nf] = __builtin_amdgcn_wmma_f32_16x16x32_f16(
            false, a1, false, bf[nf], (short)0, acc[1][nf], false, false);
      }
    }
    __syncthreads();
  }

  #pragma unroll
  for (int mi = 0; mi < 2; ++mi) {
    #pragma unroll
    for (int nf = 0; nf < 4; ++nf) {
      const int col = nBase + nf * 16 + r;
      const float bv = bias[col];
      #pragma unroll
      for (int rr = 0; rr < 8; ++rr) {
        const int row = mBlk + wave * 32 + mi * 16 + 8 * sub + rr;
        const float v = acc[mi][nf][rr] + bv;
        if (transposeOut) Cmat[(size_t)col * M + row] = (half_t)v;
        else              Cmat[(size_t)row * N + col] = (half_t)v;
      }
    }
  }
}

// -----------------------------------------------------------------------------
// Causal flash attention. Block = 4 waves = 64 query rows of one (b,h).
// K/V tiles shared across the block in LDS, async double-buffered.
// -----------------------------------------------------------------------------
__global__ __launch_bounds__(128) void attn_fmha_wmma(
    const half_t* __restrict__ Qh,  // [B*T, D]
    const half_t* __restrict__ Kh,  // [B*T, D]
    const half_t* __restrict__ Vt,  // [D, B*T]  (transposed V)
    float* __restrict__ Out,        // [B, T, D] fp32
    int T, int D, float scale)
{
  __shared__ __align__(16) half_t Ktile[2][32][64];    // 8 KB
  __shared__ __align__(16) half_t Vtile[2][64][32];    // 8 KB
  __shared__ __align__(16) half_t psh[4][16][32];      // 4 KB, per-wave P staging

  const int t    = threadIdx.x;
  const int lane = t & 31;
  const int wave = t >> 5;
  const int r    = lane & 15;
  const int sub  = lane >> 4;

  const int qBlk  = blockIdx.x * 64;
  const int qBase = qBlk + wave * 16;
  const int hoff  = blockIdx.y * 64;             // head_dim = 64
  const int b     = blockIdx.z;
  const size_t boff = (size_t)b * T * D;
  const size_t MT   = (size_t)gridDim.z * T;     // leading dim of Vt
  const size_t bT   = (size_t)b * T;

  // cooperative async tile loader: K 32x64 + V 64x32 (2+2 chunks/thread)
  auto load_tiles = [&](int j, int buf) {
    #pragma unroll
    for (int i = 0; i < 2; ++i) {
      const int c = t + 128 * i;
      const int row = c >> 3, col = (c & 7) * 8;
      g2l_async_b128(&Ktile[buf][row][col],
                     Kh + boff + (size_t)(j + row) * D + hoff + col);
    }
    #pragma unroll
    for (int i = 0; i < 2; ++i) {
      const int c = t + 128 * i;
      const int row = c >> 2, col = (c & 3) * 8;
      g2l_async_b128(&Vtile[buf][row][col],
                     Vt + (size_t)(hoff + row) * MT + bT + j + col);
    }
  };

  load_tiles(0, 0);

  // Resident Q fragments: 16x64 as two 16x32 A-fragments (overlap with async)
  const half_t* qrow = Qh + boff + (size_t)(qBase + r) * D + hoff + sub * 8;
  const v16h qa0 = load_frag16(qrow,      qrow + 16);
  const v16h qa1 = load_frag16(qrow + 32, qrow + 48);

  v8f o[4] = {};
  float mrow[8], lrow[8];
  #pragma unroll
  for (int i = 0; i < 8; ++i) { mrow[i] = -3.0e38f; lrow[i] = 0.0f; }

  const int jTop = qBlk + 63;       // last key row the block needs
  const int qEnd = qBase + 15;      // last key this wave needs
  for (int j = 0; j <= jTop; j += 32) {
    const int buf = (j >> 5) & 1;
    const bool hasNext = (j + 32 <= jTop);
    if (hasNext) { load_tiles(j + 32, buf ^ 1); WAIT_ASYNC(4); }
    else         { WAIT_ASYNC(0); }
    __syncthreads();

    if (j <= qEnd) {   // wave-uniform: skip fully masked tiles
      // ---- load K fragments (distinct regs), issue 4 S wmma back-to-back
      v16h bk0[2], bk1[2];
      #pragma unroll
      for (int nh = 0; nh < 2; ++nh) {
        const half_t* kp = &Ktile[buf][nh * 16 + r][sub * 8];
        bk0[nh] = load_frag16(kp,      kp + 16);
        bk1[nh] = load_frag16(kp + 32, kp + 48);
      }
      v8f s[2];
      #pragma unroll
      for (int nh = 0; nh < 2; ++nh) {
        v8f z = {};
        z = __builtin_amdgcn_wmma_f32_16x16x32_f16(false, qa0, false, bk0[nh], (short)0, z, false, false);
        z = __builtin_amdgcn_wmma_f32_16x16x32_f16(false, qa1, false, bk1[nh], (short)0, z, false, false);
        s[nh] = z;
      }

      // ---- load V fragments early: ds latency hides under softmax VALU
      v16h bvf[4];
      #pragma unroll
      for (int nf = 0; nf < 4; ++nf) {
        const half_t* vp = &Vtile[buf][nf * 16 + r][sub * 8];
        bvf[nf] = load_frag16(vp, vp + 16);
      }

      // ---- scale + causal mask + online softmax
      float p0v[8], p1v[8];
      #pragma unroll
      for (int rr = 0; rr < 8; ++rr) {
        const int rowq = qBase + 8 * sub + rr;
        float s0 = s[0][rr] * scale;
        float s1 = s[1][rr] * scale;
        if (j + r      > rowq) s0 = -3.0e38f;
        if (j + 16 + r > rowq) s1 = -3.0e38f;
        float tm = fmaxf(s0, s1);
        #pragma unroll
        for (int xm = 1; xm <= 8; xm <<= 1)
          tm = fmaxf(tm, __shfl_xor(tm, xm, 32));
        const float mn = fmaxf(mrow[rr], tm);
        const float p0 = __expf(s0 - mn);
        const float p1 = __expf(s1 - mn);
        p0v[rr] = p0; p1v[rr] = p1;
        float ts = p0 + p1;
        #pragma unroll
        for (int xm = 1; xm <= 8; xm <<= 1)
          ts += __shfl_xor(ts, xm, 32);
        const float alpha = __expf(mrow[rr] - mn);
        lrow[rr] = lrow[rr] * alpha + ts;
        mrow[rr] = mn;
        #pragma unroll
        for (int nf = 0; nf < 4; ++nf) o[nf][rr] = o[nf][rr] * alpha;
      }

      // ---- re-layout P: C-fragment (f32) -> LDS (f16) -> A-fragment
      #pragma unroll
      for (int rr = 0; rr < 8; ++rr) {
        psh[wave][8 * sub + rr][r]      = (half_t)p0v[rr];
        psh[wave][8 * sub + rr][16 + r] = (half_t)p1v[rr];
      }
      WAIT_DS0();
      const half_t* pl = &psh[wave][r][sub * 8];
      const v16h pa = load_frag16(pl, pl + 16);

      // ---- O += P(16x32) * V(32x64) : 4 wmma, V fragments already resident
      #pragma unroll
      for (int nf = 0; nf < 4; ++nf) {
        o[nf] = __builtin_amdgcn_wmma_f32_16x16x32_f16(
            false, pa, false, bvf[nf], (short)0, o[nf], false, false);
      }
    }
    __syncthreads();
  }

  // ---- normalize and store fp32 output
  #pragma unroll
  for (int nf = 0; nf < 4; ++nf) {
    const int col = hoff + nf * 16 + r;
    #pragma unroll
    for (int rr = 0; rr < 8; ++rr) {
      const int row = qBase + 8 * sub + rr;
      Out[boff + (size_t)row * D + col] = o[nf][rr] / lrow[rr];
    }
  }
}

// -----------------------------------------------------------------------------
extern "C" void kernel_launch(void* const* d_in, const int* in_sizes, int n_in,
                              void* d_out, int out_size, void* d_ws, size_t ws_size,
                              hipStream_t stream) {
  (void)in_sizes; (void)n_in; (void)out_size; (void)ws_size;
  const int B = 8, T = 2048, D = 1024, H = 16;
  const int M = B * T;                       // 16384 rows
  const size_t NX = (size_t)M * D;           // 16,777,216 elements
  const float scale = 0.125f;                // 1/sqrt(64)

  const float* x  = (const float*)d_in[0];
  const float* Wq = (const float*)d_in[1];
  const float* bq = (const float*)d_in[2];
  const float* Wk = (const float*)d_in[3];
  const float* bk = (const float*)d_in[4];
  const float* Wv = (const float*)d_in[5];
  const float* bv = (const float*)d_in[6];
  float* out = (float*)d_out;

  // workspace layout (f16): xh | Qh | Kh | Vt | Wh
  half_t* xh = (half_t*)d_ws;
  half_t* Qh = xh + NX;
  half_t* Kh = Qh + NX;
  half_t* Vt = Kh + NX;
  half_t* Wh = Vt + NX;

  // x -> f16
  {
    const int n4 = (int)(NX / 4);            // 4,194,304
    f32_to_f16_4<<<n4 / 256, 256, 0, stream>>>(x, xh, n4);
  }

  const dim3 ggrid(M / 128, D / 64);         // 128 x 16 workgroups
  const int wn4 = D * D / 4;                 // 262,144

  // Q = x Wq^T + bq
  f32_to_f16_4<<<wn4 / 256, 256, 0, stream>>>(Wq, Wh, wn4);
  qkv_gemm_wmma<<<ggrid, 128, 0, stream>>>(xh, Wh, bq, Qh, M, D, D, 0);
  // K = x Wk^T + bk
  f32_to_f16_4<<<wn4 / 256, 256, 0, stream>>>(Wk, Wh, wn4);
  qkv_gemm_wmma<<<ggrid, 128, 0, stream>>>(xh, Wh, bk, Kh, M, D, D, 0);
  // V = x Wv^T + bv, stored transposed [D, M] for contiguous PV B-fragments
  f32_to_f16_4<<<wn4 / 256, 256, 0, stream>>>(Wv, Wh, wn4);
  qkv_gemm_wmma<<<ggrid, 128, 0, stream>>>(xh, Wh, bv, Vt, M, D, D, 1);

  // causal flash attention
  const dim3 agrid(T / 64, H, B);            // 32 x 16 x 8
  attn_fmha_wmma<<<agrid, 128, 0, stream>>>(Qh, Kh, Vt, out, T, D, scale);
}
